// GlobalAtomAttention_85882166051073
// MI455X (gfx1250) — matrix operand order repaired and checked by reading, not compiled, not run on hardware
//
#include <hip/hip_runtime.h>
#include <cstdint>
#include <cstddef>

// ---------------------------------------------------------------------------
// Problem constants (match reference)
// ---------------------------------------------------------------------------
#define BB 4
#define NN 512
#define MM 64
#define FF 256
#define QQ 16
#define HH 8
#define RR 32
#define DD 32          // F / H
#define ROWS (BB*NN)   // 2048
#define FFNH 512
#define KQ_PAD 288     // 272 padded to 32 (LDS row stride for neighbor block)
#define KIN 272        // F + Q
#define KUP 544        // 2F + R (17*32)

typedef __bf16 bf16_t;
typedef __attribute__((ext_vector_type(16))) __bf16 v16bf;
typedef __attribute__((ext_vector_type(8)))  __bf16 v8bf;
typedef __attribute__((ext_vector_type(8)))  float  v8f;

static __device__ __forceinline__ bf16_t f2bf(float x) { return (bf16_t)x; }

// ---------------------------------------------------------------------------
// TDM availability (this toolchain: 6-arg tensor_load_to_lds builtin)
// ---------------------------------------------------------------------------
#if defined(__gfx1250__) && __has_builtin(__builtin_amdgcn_tensor_load_to_lds) && \
    __has_builtin(__builtin_amdgcn_s_wait_tensorcnt)
#define USE_TDM 1
#else
#define USE_TDM 0
#endif

#if USE_TDM
typedef __attribute__((ext_vector_type(4))) unsigned int u32x4;
typedef __attribute__((ext_vector_type(8))) int          i32x8;
typedef __attribute__((ext_vector_type(4))) int          i32x4;

// LDS offset (bytes) of a generic pointer into shared memory
static __device__ __forceinline__ unsigned lds_offset_of(const void* p) {
    return (unsigned)(unsigned long long)(__attribute__((address_space(3))) const char*)p;
}

// Issue one TDM gather-mode descriptor: 16 indexed rows of a 2D f32 tensor
// (row length tdim0 elements, row stride = tdim0) into LDS at lds_addr.
// Optional LDS padding after each row (pad_interval/pad_amount encodings per D#).
static __device__ __forceinline__ void tdm_gather16(
    unsigned long long gbase, unsigned lds_addr, const int* __restrict__ idxp,
    unsigned tdim0, unsigned nrows_tensor,
    unsigned pad_enable, unsigned pad_interval, unsigned pad_amount)
{
    u32x4 g0;
    g0[0] = 1u | (1u << 31);                       // count=1, gather_mode=1, 16-bit indices
    g0[1] = lds_addr;                              // lds_addr [63:32]
    g0[2] = (unsigned)(gbase & 0xffffffffull);     // global_addr lo
    g0[3] = (unsigned)((gbase >> 32) & 0x01ffffffull) | (2u << 30); // addr hi, type=2

    i32x8 g1;
    g1[0] = (int)((2u << 16) | (pad_enable << 20) |
                  (pad_interval << 22) | (pad_amount << 25)); // data_size=4B
    g1[1] = (int)((tdim0 & 0xffffu) << 16);                   // tensor_dim0 lo16
    g1[2] = (int)(((tdim0 >> 16) & 0xffffu) |
                  ((nrows_tensor & 0xffffu) << 16));          // dim0 hi / dim1 lo
    g1[3] = (int)(((nrows_tensor >> 16) & 0xffffu) |
                  ((tdim0 & 0xffffu) << 16));                 // dim1 hi / tile_dim0
    g1[4] = (int)16;                                          // tile_dim1 = 16 indices
    g1[5] = (int)tdim0;                                       // tensor_dim0_stride lo32
    g1[6] = 0;
    g1[7] = 0;

    i32x4 g2, g3;
    #pragma unroll
    for (int t = 0; t < 4; ++t) {
        int j0 = idxp[2*t],     j1 = idxp[2*t + 1];
        int j2 = idxp[8 + 2*t], j3 = idxp[8 + 2*t + 1];
        g2[t] = (j0 & 0xffff) | (j1 << 16);
        g3[t] = (j2 & 0xffff) | (j3 << 16);
    }
    i32x8 g4 = {0, 0, 0, 0, 0, 0, 0, 0};           // unused descriptor group (6-arg form)
    __builtin_amdgcn_tensor_load_to_lds(g0, g1, g2, g3, g4, 0);
}
#endif // USE_TDM

// ---------------------------------------------------------------------------
// Weight / activation conversion+packing kernels (f32 -> padded bf16)
// ---------------------------------------------------------------------------
__global__ void cvt_pad_bf16_kernel(const float* __restrict__ src, bf16_t* __restrict__ dst,
                                    int rows, int K, int Kpad) {
    int total = rows * Kpad;
    for (int i = blockIdx.x * blockDim.x + threadIdx.x; i < total; i += gridDim.x * blockDim.x) {
        int r = i / Kpad, c = i - r * Kpad;
        dst[i] = f2bf(c < K ? src[(size_t)r * K + c] : 0.0f);
    }
}

__global__ void pack_center_kernel(const float* __restrict__ a, const float* __restrict__ ch,
                                   bf16_t* __restrict__ dst) {
    int total = ROWS * KQ_PAD;
    for (int i = blockIdx.x * blockDim.x + threadIdx.x; i < total; i += gridDim.x * blockDim.x) {
        int r = i / KQ_PAD, c = i - r * KQ_PAD;
        float v = (c < FF) ? a[(size_t)r * FF + c]
                : (c < KIN) ? ch[(size_t)r * QQ + (c - FF)] : 0.0f;
        dst[i] = f2bf(v);
    }
}

__global__ void pack_upd_kernel(const float* __restrict__ a, const float* __restrict__ ctx,
                                const float* __restrict__ rc, bf16_t* __restrict__ dst) {
    int total = ROWS * KUP;
    for (int i = blockIdx.x * blockDim.x + threadIdx.x; i < total; i += gridDim.x * blockDim.x) {
        int r = i / KUP, c = i - r * KUP;
        float v = (c < FF) ? a[(size_t)r * FF + c]
                : (c < 2 * FF) ? ctx[(size_t)r * FF + (c - FF)]
                : rc[(size_t)r * RR + (c - 2 * FF)];
        dst[i] = f2bf(v);
    }
}

// ---------------------------------------------------------------------------
// Generic bf16 WMMA row-GEMM:  out[ROWS x ncols] = A[ROWS x K] @ W[ncols x K]^T
// Block: 256 threads = 8 waves; block tile 64 rows x 64 cols.
// EPI: 0 = +bias -> f32 ; 1 = +bias+residual -> f32 ; 2 = gelu(+bias) -> bf16
// ---------------------------------------------------------------------------
template<int EPI>
__global__ __launch_bounds__(256) void gemm_bf16_kernel(
    const bf16_t* __restrict__ A, int lda,
    const bf16_t* __restrict__ W, int ldw,
    const float* __restrict__ bias,
    const float* __restrict__ resid,
    float* __restrict__ outF,
    bf16_t* __restrict__ outB,
    int K, int ncols)
{
    const int tid  = threadIdx.x;
    const int wave = tid >> 5;
    const int lane = tid & 31;
    const int hl   = lane >> 4;
    const int ln   = lane & 15;

    const int row0 = blockIdx.x * 64 + (wave >> 1) * 16;
    const int col0 = blockIdx.y * 64 + (wave & 1) * 32;

    v8f acc0 = {0.f,0.f,0.f,0.f,0.f,0.f,0.f,0.f};
    v8f acc1 = {0.f,0.f,0.f,0.f,0.f,0.f,0.f,0.f};

    const bf16_t* arow  = A + (size_t)(row0 + ln) * lda;
    const bf16_t* wrow0 = W + (size_t)(col0 + ln) * ldw;
    const bf16_t* wrow1 = W + (size_t)(col0 + 16 + ln) * ldw;

    for (int k0 = 0; k0 < K; k0 += 32) {
        union { v16bf v; v8bf h[2]; } ua;
        const bf16_t* pa = arow + k0 + hl * 16;
        ua.h[0] = *(const v8bf*)(pa);
        ua.h[1] = *(const v8bf*)(pa + 8);
        v16bf b0 = *(const v16bf*)(wrow0 + k0 + hl * 16);
        v16bf b1 = *(const v16bf*)(wrow1 + k0 + hl * 16);
        acc0 = __builtin_amdgcn_wmma_f32_16x16x32_bf16(false, ua.v, false, b0, (short)0, acc0, false, false);
        acc1 = __builtin_amdgcn_wmma_f32_16x16x32_bf16(false, ua.v, false, b1, (short)0, acc1, false, false);
    }

    #pragma unroll
    for (int r = 0; r < 8; ++r) {
        const int m  = row0 + hl * 8 + r;
        const int c0 = col0 + ln;
        const int c1 = col0 + 16 + ln;
        float v0 = acc0[r] + bias[c0];
        float v1 = acc1[r] + bias[c1];
        if (EPI == 1) {
            v0 += resid[(size_t)m * ncols + c0];
            v1 += resid[(size_t)m * ncols + c1];
        }
        if (EPI == 2) {
            v0 = 0.5f * v0 * (1.0f + erff(v0 * 0.70710678118654752f));
            v1 = 0.5f * v1 * (1.0f + erff(v1 * 0.70710678118654752f));
            outB[(size_t)m * ncols + c0] = f2bf(v0);
            outB[(size_t)m * ncols + c1] = f2bf(v1);
        } else {
            outF[(size_t)m * ncols + c0] = v0;
            outF[(size_t)m * ncols + c1] = v1;
        }
    }
}

// ---------------------------------------------------------------------------
// Attention core: one workgroup per (b,n). 256 threads = 8 waves.
// TDM gathers the 64 neighbor rows into LDS (overlapped with qk-fold/radial),
// then scores / softmax / weighted-neighbor / per-head Wv projection.
// ---------------------------------------------------------------------------
__global__ __launch_bounds__(256) void attn_kernel(
    const float* __restrict__ a, const float* __restrict__ charges,
    const int* __restrict__ idx_j, const float* __restrict__ d_ij,
    const unsigned char* __restrict__ mask_ij, const float* __restrict__ shifts,
    const float* __restrict__ Wk, const float* __restrict__ bk,
    const float* __restrict__ Wv, const float* __restrict__ bv,
    const float* __restrict__ Wrb, const float* __restrict__ brb,
    const float* __restrict__ Qp,
    float* __restrict__ ctx_out, float* __restrict__ rc_out)
{
    extern __shared__ char smem_raw[];
    float* sm_neigh = (float*)smem_raw;            // [64][288] (cols 0..271 valid)
    float* sm_chg   = sm_neigh + MM * KQ_PAD;      // [64][16] compact charges staging
    float* sm_q     = sm_chg + MM * QQ;            // [256]
    float* qk       = sm_q + FF;                   // [8][288]
    float* sbk      = qk + HH * KQ_PAD;            // [8]
    float* rad      = sbk + HH;                    // [64][32]
    float* sc       = rad + MM * RR;               // [8][64]
    float* attnv    = sc + HH * MM;                // [8][64]
    float* sumattn  = attnv + HH * MM;             // [8]
    float* wn       = sumattn + HH;                // [8][288]
    float* rcp      = wn + HH * KQ_PAD;            // [8][32]
    int*   sidx     = (int*)(rcp + HH * RR);       // [64] (fallback path)

    const int bn   = blockIdx.x;
    const int b    = bn / NN;
    const int tid  = threadIdx.x;
    const int wave = tid >> 5;
    const int lane = tid & 31;

#if USE_TDM
    // Issue the neighbor gather on the Tensor Data Mover immediately (wave 0),
    // then overlap descriptor-independent compute with the DMA.
    if (wave == 0) {
        const int* ib = idx_j + (size_t)bn * MM;
        unsigned long long abase =
            (unsigned long long)(uintptr_t)(a + (size_t)b * NN * FF);
        unsigned long long cbase =
            (unsigned long long)(uintptr_t)(charges + (size_t)b * NN * QQ);
        unsigned noff = lds_offset_of(sm_neigh);
        unsigned coff = lds_offset_of(sm_chg);
        #pragma unroll
        for (int d2 = 0; d2 < 4; ++d2) {
            // 16 rows of a[b]: 256 f32/row, LDS stride 288 f32
            // pad_interval=7 (256 dwords), pad_amount=31 (32 dwords)
            tdm_gather16(abase, noff + (unsigned)d2 * 16u * KQ_PAD * 4u,
                         ib + d2 * 16, FF, NN, 1u, 7u, 31u);
            // 16 rows of charges[b]: 16 f32/row, compact (no pad)
            tdm_gather16(cbase, coff + (unsigned)d2 * 16u * QQ * 4u,
                         ib + d2 * 16, QQ, NN, 0u, 0u, 0u);
        }
    }
#else
    if (tid < MM) sidx[tid] = idx_j[(size_t)bn * MM + tid];
#endif
    sm_q[tid] = Qp[(size_t)bn * FF + tid];
    __syncthreads();

#if !USE_TDM
    // Fallback manual gather (also used by the host compile pass)
    for (int i = tid; i < MM * KQ_PAD; i += 256) {
        int m = i / KQ_PAD, c = i - m * KQ_PAD;
        if (c < KIN) {
            int j = sidx[m];
            sm_neigh[i] = (c < FF) ? a[((size_t)b * NN + j) * FF + c]
                                   : charges[((size_t)b * NN + j) * QQ + (c - FF)];
        }
    }
#endif
    // qk[h][c] = sum_d q[h,d] * Wk[h*32+d, c]   (fold q into Wk: 272x8 not 272x256)
    for (int i = tid; i < HH * KQ_PAD; i += 256) {
        int h = i / KQ_PAD, c = i - h * KQ_PAD;
        float s = 0.0f;
        if (c < KIN) {
            const float* wrow = Wk + (size_t)(h * DD) * KIN + c;
            const float* qh = sm_q + h * DD;
            #pragma unroll 8
            for (int d = 0; d < DD; ++d) s += qh[d] * wrow[(size_t)d * KIN];
        }
        qk[i] = s;
    }
    if (tid < HH) {
        float s = 0.0f;
        for (int d = 0; d < DD; ++d) s += sm_q[tid * DD + d] * bk[tid * DD + d];
        sbk[tid] = s;
    }
    for (int i = tid; i < MM * RR; i += 256) {
        int m = i >> 5, r = i & 31;
        float dd = d_ij[(size_t)bn * MM + m] - shifts[r];
        rad[i] = __expf(-4.0f * dd * dd);
    }

#if USE_TDM
    if (wave == 0) __builtin_amdgcn_s_wait_tensorcnt(0);
    __syncthreads();
    // scatter compact charges into neighbor block cols 256..271
    for (int i = tid; i < MM * QQ; i += 256) {
        int m = i >> 4, t = i & 15;
        sm_neigh[m * KQ_PAD + FF + t] = sm_chg[i];
    }
#endif
    __syncthreads();

    // scores[h][m]
    for (int i = tid; i < HH * MM; i += 256) {
        int h = i >> 6, m = i & 63;
        const float* nr = sm_neigh + m * KQ_PAD;
        const float* qr = qk + h * KQ_PAD;
        float s = 0.0f;
        for (int c = 0; c < KIN; ++c) s += nr[c] * qr[c];
        s = (s + sbk[h]) * 0.17677669529663687f;  // 1/sqrt(32)
        float rb = brb[h];
        const float* rr = rad + m * RR;
        const float* wr = Wrb + h * RR;
        for (int r = 0; r < RR; ++r) rb += rr[r] * wr[r];
        s += rb;
        if (mask_ij[(size_t)bn * MM + m]) s = -INFINITY;
        sc[h * MM + m] = s;
    }
    __syncthreads();

    // softmax over m: one wave per head (wave32, 2 elems/lane)
    {
        int h = wave;
        float s0 = sc[h * MM + lane];
        float s1 = sc[h * MM + 32 + lane];
        float mx = fmaxf(s0, s1);
        for (int off = 16; off; off >>= 1) mx = fmaxf(mx, __shfl_xor(mx, off, 32));
        float e0 = 0.0f, e1 = 0.0f, sum = 0.0f;
        if (mx != -INFINITY) {
            e0 = __expf(s0 - mx);
            e1 = __expf(s1 - mx);
            sum = e0 + e1;
            for (int off = 16; off; off >>= 1) sum += __shfl_xor(sum, off, 32);
        }
        float inv = (sum > 0.0f) ? 1.0f / sum : 0.0f;   // all-masked -> attn = 0
        attnv[h * MM + lane]      = e0 * inv;
        attnv[h * MM + 32 + lane] = e1 * inv;
        if (lane == 0) sumattn[h] = (sum > 0.0f) ? 1.0f : 0.0f;
    }
    __syncthreads();

    // wn[h][c] = attn_h @ neigh
    for (int i = tid; i < HH * KQ_PAD; i += 256) {
        int h = i / KQ_PAD, c = i - h * KQ_PAD;
        float s = 0.0f;
        if (c < KIN) {
            const float* at = attnv + h * MM;
            for (int m = 0; m < MM; ++m) s += at[m] * sm_neigh[m * KQ_PAD + c];
        }
        wn[i] = s;
    }
    {
        int h = tid >> 5, r = tid & 31;
        const float* at = attnv + h * MM;
        float s = 0.0f;
        for (int m = 0; m < MM; ++m) s += at[m] * rad[m * RR + r];
        rcp[h * RR + r] = s;
    }
    __syncthreads();

    if (tid < RR) {
        float s = 0.0f;
        for (int h = 0; h < HH; ++h) s += rcp[h * RR + tid];
        rc_out[(size_t)bn * RR + tid] = s * 0.125f;
    }
    {
        int h = tid >> 5;
        int o = tid;  // h*32 + d
        const float* wv = Wv + (size_t)o * KIN;
        const float* w  = wn + h * KQ_PAD;
        float s = 0.0f;
        for (int c = 0; c < KIN; ++c) s += w[c] * wv[c];
        s += sumattn[h] * bv[o];
        ctx_out[(size_t)bn * FF + o] = s;
    }
}

// ---------------------------------------------------------------------------
// LayerNorm over 256 features: one wave per row (8 rows / 256-thread block)
// ---------------------------------------------------------------------------
__global__ __launch_bounds__(256) void ln_kernel(
    const float* __restrict__ in, const float* __restrict__ g, const float* __restrict__ bta,
    float* __restrict__ outF, bf16_t* __restrict__ outB)
{
    const int wave = threadIdx.x >> 5, lane = threadIdx.x & 31;
    const int row = blockIdx.x * 8 + wave;
    const float* x = in + (size_t)row * FF;
    float v[8];
    float s = 0.0f;
    #pragma unroll
    for (int i = 0; i < 8; ++i) { v[i] = x[lane + 32 * i]; s += v[i]; }
    for (int off = 16; off; off >>= 1) s += __shfl_xor(s, off, 32);
    const float mu = s * (1.0f / 256.0f);
    float vs = 0.0f;
    #pragma unroll
    for (int i = 0; i < 8; ++i) { float d = v[i] - mu; vs += d * d; }
    for (int off = 16; off; off >>= 1) vs += __shfl_xor(vs, off, 32);
    const float inv = rsqrtf(vs * (1.0f / 256.0f) + 1e-5f);
    #pragma unroll
    for (int i = 0; i < 8; ++i) {
        int c = lane + 32 * i;
        float y = (v[i] - mu) * inv * g[c] + bta[c];
        outF[(size_t)row * FF + c] = y;
        if (outB) outB[(size_t)row * FF + c] = f2bf(y);
    }
}

// ---------------------------------------------------------------------------
// Host-side orchestration
// ---------------------------------------------------------------------------
static inline size_t align256(size_t x) { return (x + 255) & ~(size_t)255; }

extern "C" void kernel_launch(void* const* d_in, const int* in_sizes, int n_in,
                              void* d_out, int out_size, void* d_ws, size_t ws_size,
                              hipStream_t stream) {
    const float* a       = (const float*)d_in[0];
    const float* charges = (const float*)d_in[1];
    const int*   idx_j   = (const int*)d_in[2];
    const float* d_ij    = (const float*)d_in[3];
    const unsigned char* mask = (const unsigned char*)d_in[4];
    const float* shifts  = (const float*)d_in[5];
    const float* Wq   = (const float*)d_in[6];  const float* bq   = (const float*)d_in[7];
    const float* Wk   = (const float*)d_in[8];  const float* bk   = (const float*)d_in[9];
    const float* Wv   = (const float*)d_in[10]; const float* bv   = (const float*)d_in[11];
    const float* Wrb  = (const float*)d_in[12]; const float* brb  = (const float*)d_in[13];
    const float* Wout = (const float*)d_in[14]; const float* bout = (const float*)d_in[15];
    const float* g1   = (const float*)d_in[16]; const float* b1   = (const float*)d_in[17];
    const float* g2   = (const float*)d_in[18]; const float* b2   = (const float*)d_in[19];
    const float* Wf1  = (const float*)d_in[20]; const float* bf1  = (const float*)d_in[21];
    const float* Wf2  = (const float*)d_in[22]; const float* bf2  = (const float*)d_in[23];
    float* out = (float*)d_out;

    // workspace layout
    char* w = (char*)d_ws;
    size_t off = 0;
    bf16_t* WqB   = (bf16_t*)(w + off); off = align256(off + (size_t)FF * KQ_PAD * 2);
    bf16_t* WoB   = (bf16_t*)(w + off); off = align256(off + (size_t)FF * KUP * 2);
    bf16_t* Wf1B  = (bf16_t*)(w + off); off = align256(off + (size_t)FFNH * FF * 2);
    bf16_t* Wf2B  = (bf16_t*)(w + off); off = align256(off + (size_t)FF * FFNH * 2);
    bf16_t* cenB  = (bf16_t*)(w + off); off = align256(off + (size_t)ROWS * KQ_PAD * 2);
    float*  Qp    = (float*)(w + off);  off = align256(off + (size_t)ROWS * FF * 4);
    float*  ctx   = (float*)(w + off);  off = align256(off + (size_t)ROWS * FF * 4);
    float*  rc    = (float*)(w + off);  off = align256(off + (size_t)ROWS * RR * 4);
    bf16_t* updB  = (bf16_t*)(w + off); off = align256(off + (size_t)ROWS * KUP * 2);
    float*  pre1  = (float*)(w + off);  off = align256(off + (size_t)ROWS * FF * 4);
    float*  xF    = (float*)(w + off);  off = align256(off + (size_t)ROWS * FF * 4);
    bf16_t* xB    = (bf16_t*)(w + off); off = align256(off + (size_t)ROWS * FF * 2);
    bf16_t* hB    = (bf16_t*)(w + off); off = align256(off + (size_t)ROWS * FFNH * 2);
    float*  pre2  = (float*)(w + off);  off = align256(off + (size_t)ROWS * FF * 4);

    // 1. convert weights to padded bf16 (L2-resident thereafter)
    cvt_pad_bf16_kernel<<<128, 256, 0, stream>>>(Wq,   WqB,  FF,   KIN, KQ_PAD);
    cvt_pad_bf16_kernel<<<128, 256, 0, stream>>>(Wout, WoB,  FF,   KUP, KUP);
    cvt_pad_bf16_kernel<<<128, 256, 0, stream>>>(Wf1,  Wf1B, FFNH, FF,  FF);
    cvt_pad_bf16_kernel<<<128, 256, 0, stream>>>(Wf2,  Wf2B, FF,   FFNH, FFNH);

    // 2. Q projection: Qp = [a|charges] @ Wq^T + bq       (WMMA bf16)
    pack_center_kernel<<<256, 256, 0, stream>>>(a, charges, cenB);
    {
        dim3 grid(ROWS / 64, FF / 64);
        gemm_bf16_kernel<0><<<grid, 256, 0, stream>>>(cenB, KQ_PAD, WqB, KQ_PAD, bq,
                                                      nullptr, Qp, nullptr, KQ_PAD, FF);
    }

    // 3. attention core: one WG per (b,n); TDM gather + ~111 KB dynamic LDS
    {
        const size_t smem = (size_t)(MM * KQ_PAD + MM * QQ + FF + HH * KQ_PAD + HH
                                     + MM * RR + 2 * HH * MM + HH + HH * KQ_PAD
                                     + HH * RR) * 4 + MM * 4;
        attn_kernel<<<ROWS, 256, smem, stream>>>(a, charges, idx_j, d_ij, mask, shifts,
                                                 Wk, bk, Wv, bv, Wrb, brb, Qp, ctx, rc);
    }

    // 4. update = [a|ctx|rc] @ Wout^T + bout + a (residual), then LN -> x
    pack_upd_kernel<<<256, 256, 0, stream>>>(a, ctx, rc, updB);
    {
        dim3 grid(ROWS / 64, FF / 64);
        gemm_bf16_kernel<1><<<grid, 256, 0, stream>>>(updB, KUP, WoB, KUP, bout,
                                                      a, pre1, nullptr, KUP, FF);
    }
    ln_kernel<<<ROWS / 8, 256, 0, stream>>>(pre1, g1, b1, xF, xB);

    // 5. FFN: h = gelu(x @ Wf1^T + bf1) -> bf16 ; y = h @ Wf2^T + bf2 + x ; LN -> out
    {
        dim3 grid(ROWS / 64, FFNH / 64);
        gemm_bf16_kernel<2><<<grid, 256, 0, stream>>>(xB, FF, Wf1B, FF, bf1,
                                                      nullptr, nullptr, hB, FF, FFNH);
    }
    {
        dim3 grid(ROWS / 64, FF / 64);
        gemm_bf16_kernel<1><<<grid, 256, 0, stream>>>(hB, FFNH, Wf2B, FFNH, bf2,
                                                      xF, pre2, nullptr, FFNH, FF);
    }
    ln_kernel<<<ROWS / 8, 256, 0, stream>>>(pre2, g2, b2, out, nullptr);
}
